// Attention_78847009620267
// MI455X (gfx1250) — matrix operand order repaired and checked by reading, not compiled, not run on hardware
//
#include <hip/hip_runtime.h>
#include <hip/hip_bf16.h>
#include <stdint.h>

typedef __bf16 bf16_t;
typedef __attribute__((ext_vector_type(16))) __bf16 v16bf;
typedef __attribute__((ext_vector_type(8)))  __bf16 v8bf;
typedef __attribute__((ext_vector_type(8)))  float   v8f;
typedef __attribute__((ext_vector_type(4)))  float   v4f;

#define LDS_STRIDE 40   // bf16 elems per LDS row: 80B (16B aligned, conflict-sparse)

union FragBF16 { v16bf v; v8bf h[2]; };

// -------- CDNA5 async global->LDS copy (ASYNCcnt-tracked) via inline asm ----
// global_load_async_to_lds_b128: VDST = per-lane LDS byte offset,
// VADDR = 64-bit global address (GV mode, saddr=off).  ISA §15.18.3 op 98.
#if defined(__HIP_DEVICE_COMPILE__)
#define HAVE_ASYNC_LDS 1
#else
#define HAVE_ASYNC_LDS 0
#endif

__device__ __forceinline__ void async_copy_b128(const void* g, void* l) {
#if HAVE_ASYNC_LDS
    unsigned long long gaddr = (unsigned long long)(uintptr_t)g;
    unsigned lds_off = (unsigned)(uintptr_t)l;   // flat->LDS: low 32 bits
    asm volatile("global_load_async_to_lds_b128 %0, %1, off"
                 :: "v"(lds_off), "v"(gaddr) : "memory");
#else
    *(v8bf*)l = *(const v8bf*)g;
#endif
}

__device__ __forceinline__ void wait_async_lds() {
#if HAVE_ASYNC_LDS
    asm volatile("s_wait_asynccnt 0" ::: "memory");
#endif
}

// ---------------------------------------------------------------------------
// fp32 [rows][cols] -> bf16 [cols][rows]  (weight transpose + downconvert)
// ---------------------------------------------------------------------------
__global__ __launch_bounds__(256) void transpose_to_bf16(
    const float* __restrict__ src, bf16_t* __restrict__ dst, int rows, int cols)
{
    size_t idx = (size_t)blockIdx.x * 256 + threadIdx.x;
    if (idx >= (size_t)rows * cols) return;
    int r = (int)(idx / cols);
    int c = (int)(idx % cols);
    dst[(size_t)c * rows + r] = (bf16_t)src[idx];
}

// ---------------------------------------------------------------------------
// hs = hidden @ W_h + b_h   ([64,512]x[512,512], trivial FLOPs)
// ---------------------------------------------------------------------------
__global__ __launch_bounds__(256) void hs_kernel(
    const float* __restrict__ hidden, const float* __restrict__ Wh,
    const float* __restrict__ bh, float* __restrict__ hs, int I_)
{
    int i = blockIdx.x * 256 + threadIdx.x;
    int b = blockIdx.y;
    float acc = bh[i];
    const float* hrow = hidden + (size_t)b * I_;
    for (int k = 0; k < I_; ++k)
        acc = fmaf(hrow[k], Wh[(size_t)k * I_ + i], acc);
    hs[(size_t)b * I_ + i] = acc;
}

// ---------------------------------------------------------------------------
// WMMA GEMM: C[M,N] = A[M,K] (f32 or bf16, row major) x Bt[N,K]^T (bf16)
//   FUSE_HS: C bf16, += bias[n] + hs[m/196][n]
//   else   : C f32,  += bias[n]
// Block tile 128x64, BK=32, 256 threads = 8 waves, each wave a 32x32 tile.
// Double-buffered LDS; bf16 operands stream global->LDS via async copies.
// ---------------------------------------------------------------------------
template<bool A_IS_F32, bool FUSE_HS>
__global__ __launch_bounds__(256) void wmma_gemm(
    const void* __restrict__ Aptr,
    const bf16_t* __restrict__ Bt,
    const float* __restrict__ bias,
    const float* __restrict__ hs,
    void* __restrict__ Cptr,
    int M, int N, int K)
{
    __shared__ bf16_t As[2][128 * LDS_STRIDE];
    __shared__ bf16_t Bs[2][64 * LDS_STRIDE];

    const int tid    = threadIdx.x;
    const int lane   = tid & 31;
    const int wave   = tid >> 5;
    const int wm     = wave >> 1;     // 0..3 : 32-row strip
    const int wn     = wave & 1;      // 0..1 : 32-col strip
    const int blockM = blockIdx.x * 128;
    const int blockN = blockIdx.y * 64;
    const int l16    = lane & 15;
    const int half   = lane >> 4;

    // staging thread maps
    const int ar   = tid >> 3;         // f32 A : 32 rows/pass, 4 floats each
    const int ac   = (tid & 7) * 4;
    const int a16r = tid >> 2;         // bf16 A: 64 rows/pass, 8 bf16 each
    const int a16c = (tid & 3) * 8;
    const int br   = tid >> 2;         // B     : 64 rows, 8 bf16 each
    const int bc   = (tid & 3) * 8;

    const float*  A32 = (const float*)Aptr;
    const bf16_t* A16 = (const bf16_t*)Aptr;

    v8f acc[2][2] = {};
    const int nIter = K >> 5;

    // ---- prologue: stage tile 0 into buffer 0 ----
    if (A_IS_F32) {
#pragma unroll
        for (int i = 0; i < 4; ++i) {
            int row = ar + i * 32;
            v4f x = *(const v4f*)(A32 + (size_t)(blockM + row) * K + ac);
            bf16_t* d = &As[0][row * LDS_STRIDE + ac];
            d[0] = (bf16_t)x.x; d[1] = (bf16_t)x.y;
            d[2] = (bf16_t)x.z; d[3] = (bf16_t)x.w;
        }
    } else {
#pragma unroll
        for (int i = 0; i < 2; ++i) {
            int row = a16r + i * 64;
            async_copy_b128(A16 + (size_t)(blockM + row) * K + a16c,
                            &As[0][row * LDS_STRIDE + a16c]);
        }
    }
    async_copy_b128(Bt + (size_t)(blockN + br) * K + bc,
                    &Bs[0][br * LDS_STRIDE + bc]);

    for (int it = 0; it < nIter; ++it) {
        const int  kt       = it << 5;
        const int  cur      = it & 1;
        const int  nxt      = cur ^ 1;
        const bool has_next = (it + 1) < nIter;

        wait_async_lds();      // my async writes into [cur] done
        __syncthreads();       // everyone's done; [nxt] free to overwrite

        // ---- issue next-tile loads before compute ----
        v4f areg[4];
        if (has_next) {
            const int ktn = kt + 32;
            if (A_IS_F32) {
#pragma unroll
                for (int i = 0; i < 4; ++i)
                    areg[i] = *(const v4f*)(A32 + (size_t)(blockM + ar + i * 32) * K + ktn + ac);
            } else {
#pragma unroll
                for (int i = 0; i < 2; ++i) {
                    int row = a16r + i * 64;
                    async_copy_b128(A16 + (size_t)(blockM + row) * K + ktn + a16c,
                                    &As[nxt][row * LDS_STRIDE + a16c]);
                }
            }
            async_copy_b128(Bt + (size_t)(blockN + br) * K + ktn + bc,
                            &Bs[nxt][br * LDS_STRIDE + bc]);
        }

        // ---- fragments per ISA 16-bit layouts, 4 WMMAs on [cur] ----
        FragBF16 a[2], b[2];
#pragma unroll
        for (int mi = 0; mi < 2; ++mi) {
            int row = wm * 32 + mi * 16 + l16;
            const bf16_t* p = &As[cur][row * LDS_STRIDE + half * 8];
            a[mi].h[0] = *(const v8bf*)(p);        // K 0-7  / 8-15
            a[mi].h[1] = *(const v8bf*)(p + 16);   // K 16-23/ 24-31
        }
#pragma unroll
        for (int ni = 0; ni < 2; ++ni) {
            int nrow = wn * 32 + ni * 16 + l16;
            const bf16_t* p = &Bs[cur][nrow * LDS_STRIDE + half * 8];
            b[ni].h[0] = *(const v8bf*)(p);        // K 0-7  / 8-15
            b[ni].h[1] = *(const v8bf*)(p + 16);   // K 16-23/ 24-31
        }
#pragma unroll
        for (int mi = 0; mi < 2; ++mi)
#pragma unroll
            for (int ni = 0; ni < 2; ++ni)
                acc[mi][ni] = __builtin_amdgcn_wmma_f32_16x16x32_bf16(
                    false, a[mi].v, false, b[ni].v,
                    (short)0, acc[mi][ni], false, false);

        // ---- finish f32 A staging (convert + store) into next buffer ----
        if (A_IS_F32 && has_next) {
#pragma unroll
            for (int i = 0; i < 4; ++i) {
                bf16_t* d = &As[nxt][(ar + i * 32) * LDS_STRIDE + ac];
                d[0] = (bf16_t)areg[i].x; d[1] = (bf16_t)areg[i].y;
                d[2] = (bf16_t)areg[i].z; d[3] = (bf16_t)areg[i].w;
            }
        }
    }

    // ---- epilogue ----
#pragma unroll
    for (int mi = 0; mi < 2; ++mi) {
#pragma unroll
        for (int ni = 0; ni < 2; ++ni) {
            int gcol = blockN + wn * 32 + ni * 16 + l16;
            float bn = bias[gcol];
#pragma unroll
            for (int r = 0; r < 8; ++r) {
                int grow = blockM + wm * 32 + mi * 16 + r + half * 8;
                float v = acc[mi][ni][r] + bn;
                if (FUSE_HS) {
                    int bidx = grow / 196;  // batch index of this row
                    v += hs[(size_t)bidx * N + gcol];
                    ((bf16_t*)Cptr)[(size_t)grow * N + gcol] = (bf16_t)v;
                } else {
                    ((float*)Cptr)[(size_t)grow * N + gcol] = v;
                }
            }
        }
    }
}

// ---------------------------------------------------------------------------
// In-place softmax over L (per (b,d) column) + z = sum_l alpha*fv
// ---------------------------------------------------------------------------
__global__ __launch_bounds__(256) void softmax_z_kernel(
    const float* __restrict__ fv,   // [B,L,D]
    float* __restrict__ e_alpha,    // [B,L,D]  e in, alpha out (in place)
    float* __restrict__ z,          // [B,D]
    int L_, int D_)
{
    int d = blockIdx.x * 256 + threadIdx.x;
    int b = blockIdx.y;
    size_t base = ((size_t)b * L_) * D_ + d;

    float m = -__builtin_inff();
    float s = 0.f;
    for (int l = 0; l < L_; ++l) {
        float x  = e_alpha[base + (size_t)l * D_];
        float mn = fmaxf(m, x);
        s = s * __expf(m - mn) + __expf(x - mn);
        m = mn;
    }
    float inv = 1.0f / s;
    float acc = 0.f;
    for (int l = 0; l < L_; ++l) {
        size_t idx = base + (size_t)l * D_;
        float a = __expf(e_alpha[idx] - m) * inv;
        e_alpha[idx] = a;
        acc = fmaf(a, fv[idx], acc);
    }
    z[(size_t)b * D_ + d] = acc;
}

// ---------------------------------------------------------------------------
extern "C" void kernel_launch(void* const* d_in, const int* in_sizes, int n_in,
                              void* d_out, int out_size, void* d_ws, size_t ws_size,
                              hipStream_t stream)
{
    const float* fv     = (const float*)d_in[0];   // [B,L,D]
    const float* hidden = (const float*)d_in[1];   // [B,I]
    const float* Wf     = (const float*)d_in[2];   // [D,I]
    const float* bf     = (const float*)d_in[3];   // [I]
    const float* Wh     = (const float*)d_in[4];   // [I,I]
    const float* bh     = (const float*)d_in[5];   // [I]
    const float* Wa     = (const float*)d_in[6];   // [I,D]
    const float* ba     = (const float*)d_in[7];   // [D]

    const int B = 64, L = 196, D = 2048, I = 512;
    const int M = B * L;                           // 12544

    // workspace layout (bytes)
    char* ws = (char*)d_ws;
    bf16_t* WfT = (bf16_t*)(ws + 0);               // [I][D] bf16, 2 MB
    bf16_t* WaT = (bf16_t*)(ws + (size_t)2097152); // [D][I] bf16, 2 MB
    float*  hs  = (float*) (ws + (size_t)4194304); // [B][I] f32, 128 KB
    bf16_t* C1  = (bf16_t*)(ws + (size_t)4325376); // [M][I] bf16, 12.25 MB

    float* z       = (float*)d_out;                // [B,D]
    float* e_alpha = z + (size_t)B * D;            // [B,L,D] : e, then alpha in place

    // 1) weight downconvert + transpose (one-time, tiny)
    transpose_to_bf16<<<(D * I + 255) / 256, 256, 0, stream>>>(Wf, WfT, D, I);
    transpose_to_bf16<<<(I * D + 255) / 256, 256, 0, stream>>>(Wa, WaT, I, D);

    // 2) hidden-state projection (tiny)
    hs_kernel<<<dim3(I / 256, B), 256, 0, stream>>>(hidden, Wh, bh, hs, I);

    // 3) GEMM1: C1 = bf16(fv @ W_f + b_f + hs[b])        [12544 x 512]
    wmma_gemm<true, true><<<dim3(M / 128, I / 64), 256, 0, stream>>>(
        (const void*)fv, WfT, bf, hs, (void*)C1, M, I, D);

    // 4) GEMM2: e = C1 @ W_a + b_a  -> alpha slot of out [12544 x 2048]
    wmma_gemm<false, false><<<dim3(M / 128, D / 64), 256, 0, stream>>>(
        (const void*)C1, WaT, ba, nullptr, (void*)e_alpha, M, D, I);

    // 5) softmax over L (in place) + z
    softmax_z_kernel<<<dim3(D / 256, B), 256, 0, stream>>>(fv, e_alpha, z, L, D);
}
